// RWKV_Tmix_x070_15582141350736
// MI455X (gfx1250) — compile-verified, matrix operand order checked by reading, not run on hardware
//
#include <hip/hip_runtime.h>
#include <hip/hip_bf16.h>
#include <cstdint>

// ---------------------------------------------------------------------------
// RWKV7 Tmix forward for MI455X (gfx1250, wave32, WMMA + async-LDS DMA).
// B=4 T=1024 C=2048 H=32 N=64
// ---------------------------------------------------------------------------

#define B_  4
#define T_  1024
#define C_  2048
#define H_  32
#define N_  64
#define BT_ (B_ * T_)
#define EPSV 6.4e-4f   // 1e-5 * 8^2

typedef _Float16 h16;
typedef __attribute__((ext_vector_type(16))) _Float16 v16h;
typedef __attribute__((ext_vector_type(8)))  _Float16 v8h;
typedef __attribute__((ext_vector_type(8)))  float    v8f;

__device__ __forceinline__ float sigm(float x) { return 1.0f / (1.0f + expf(-x)); }
__device__ __forceinline__ float softplusf(float z) {
    return fmaxf(z, 0.0f) + log1pf(expf(-fabsf(z)));
}

// gfx1250 async global->LDS DMA (ASYNCcnt-tracked, no VGPR staging).
// lds_off: LDS byte offset (low 32 bits of the generic shared-pointer; the
// LDS aperture maps addr[31:0] to the LDS offset, ISA 10.2).
__device__ __forceinline__ void async_ld_b128(unsigned lds_off, const h16* gptr) {
    asm volatile("global_load_async_to_lds_b128 %0, %1, off"
                 :: "v"(lds_off), "v"(gptr)
                 : "memory");
}
__device__ __forceinline__ void wait_async0() {
    asm volatile("s_wait_asynccnt 0" ::: "memory");
}

// ---------------------------------------------------------------------------
// Weight f32 -> f16 conversion (one-shot per launch, deterministic)
// ---------------------------------------------------------------------------
__global__ void f32_to_f16_kernel(const float* __restrict__ src,
                                  h16* __restrict__ dst, size_t n) {
    size_t i = (size_t)blockIdx.x * blockDim.x + threadIdx.x;
    if (i < n) dst[i] = (h16)src[i];
}

// ---------------------------------------------------------------------------
// Token shift + six mixes + LoRA MLPs (w / a / v-gate / g).
// One block (256 threads) per (b,t) row. Also emits v_first copy and lx.
// ---------------------------------------------------------------------------
__global__ __launch_bounds__(256) void mix_mlp_kernel(
    const float* __restrict__ x,  const float* __restrict__ v_first,
    const float* __restrict__ shift_state,
    const float* __restrict__ xr_m, const float* __restrict__ xw_m,
    const float* __restrict__ xk_m, const float* __restrict__ xv_m,
    const float* __restrict__ xa_m, const float* __restrict__ xg_m,
    const float* __restrict__ w0, const float* __restrict__ w1, const float* __restrict__ w2,
    const float* __restrict__ a0, const float* __restrict__ a1, const float* __restrict__ a2,
    const float* __restrict__ v0, const float* __restrict__ v1, const float* __restrict__ v2,
    const float* __restrict__ g1, const float* __restrict__ g2,
    h16* __restrict__ Xr, h16* __restrict__ Xk, h16* __restrict__ Xv, h16* __restrict__ Xg,
    float* __restrict__ EW, float* __restrict__ Asig,
    float* __restrict__ Vgate, float* __restrict__ Gout,
    float* __restrict__ vfirst_out, float* __restrict__ lx_out)
{
    __shared__ float sxw[C_], sxa[C_], sxv[C_], sxg[C_];
    __shared__ float shw[64], sha[64], shv[32], shg[128];

    const int tid = threadIdx.x;
    const int row = blockIdx.x;          // b*T + t
    const int b   = row / T_;
    const int t   = row % T_;
    const size_t rbase = (size_t)row * C_;

    for (int c = tid; c < C_; c += 256) {
        float xc   = x[rbase + c];
        float prev = (t == 0) ? shift_state[(size_t)b * C_ + c]
                              : x[rbase - C_ + c];
        float xx = prev - xc;
        Xr[rbase + c] = (h16)(xc + xx * xr_m[c]);
        Xk[rbase + c] = (h16)(xc + xx * xk_m[c]);
        Xv[rbase + c] = (h16)(xc + xx * xv_m[c]);
        Xg[rbase + c] = (h16)(xc + xx * xg_m[c]);
        sxw[c] = xc + xx * xw_m[c];
        sxa[c] = xc + xx * xa_m[c];
        sxv[c] = xc + xx * xv_m[c];
        sxg[c] = xc + xx * xg_m[c];
        vfirst_out[rbase + c] = v_first[rbase + c];
        if (t == T_ - 1) lx_out[(size_t)b * C_ + c] = xc;
    }
    __syncthreads();

    // Hidden projections of the LoRA branches
    if (tid < 64) {                               // hw = tanh(xw @ w1)
        float acc = 0.f;
        for (int k = 0; k < C_; ++k) acc = fmaf(sxw[k], w1[k * 64 + tid], acc);
        shw[tid] = tanhf(acc);
    }
    if (tid >= 64 && tid < 128) {                 // ha = xa @ a1
        int j = tid - 64; float acc = 0.f;
        for (int k = 0; k < C_; ++k) acc = fmaf(sxa[k], a1[k * 64 + j], acc);
        sha[j] = acc;
    }
    if (tid >= 128) {                             // hg = sigmoid(xg @ g1)
        int j = tid - 128; float acc = 0.f;
        for (int k = 0; k < C_; ++k) acc = fmaf(sxg[k], g1[k * 128 + j], acc);
        shg[j] = sigm(acc);
    }
    if (tid < 32) {                               // hv = xv @ v1
        float acc = 0.f;
        for (int k = 0; k < C_; ++k) acc = fmaf(sxv[k], v1[k * 32 + tid], acc);
        shv[tid] = acc;
    }
    __syncthreads();

    // Output projections
    for (int c = tid; c < C_; c += 256) {
        float accw = 0.f, acca = 0.f, accg = 0.f, accv = 0.f;
        for (int j = 0; j < 64; ++j)  accw = fmaf(shw[j], w2[j * C_ + c], accw);
        for (int j = 0; j < 64; ++j)  acca = fmaf(sha[j], a2[j * C_ + c], acca);
        for (int j = 0; j < 128; ++j) accg = fmaf(shg[j], g2[j * C_ + c], accg);
        for (int j = 0; j < 32; ++j)  accv = fmaf(shv[j], v2[j * C_ + c], accv);
        float wv = w0[c] + accw;
        float w_ = -softplusf(-wv) - 0.5f;        // decay
        EW[rbase + c]    = expf(w_);              // store exp(w) for the scan
        Asig[rbase + c]  = sigm(a0[c] + acca);
        Vgate[rbase + c] = sigm(v0[c] + accv);
        Gout[rbase + c]  = accg;
    }
}

// ---------------------------------------------------------------------------
// f16 WMMA GEMM: Cmat[M x Nn] = A[M x K] * Wt[Nn x K]^T   (f32 accumulate)
// Block tile 128(M) x 256(N), K-step 32, double-buffered LDS filled by
// global_load_async_to_lds_b128.  8 waves; each wave owns a 64x64 patch
// (4x4 grid of 16x16 WMMA tiles -> 16 v_wmma per K-step).
// ---------------------------------------------------------------------------
__global__ __launch_bounds__(256) void gemm_f16_wmma(
    const h16* __restrict__ A, const h16* __restrict__ Wt,
    float* __restrict__ Cmat, int M, int Nn, int K)
{
    __shared__ __align__(128) h16 As[2][128][32];   // 2 x 8 KB
    __shared__ __align__(128) h16 Bs[2][256][32];   // 2 x 16 KB

    const int tid  = threadIdx.x;
    const int lane = tid & 31;
    const int wave = tid >> 5;       // 0..7
    const int wm   = wave >> 2;      // 0..1  -> 64-row strip
    const int wn   = wave & 3;       // 0..3  -> 64-col strip
    const int m0 = blockIdx.y * 128;
    const int n0 = blockIdx.x * 256;

    const int lr16  = lane & 15;
    const int khalf = (lane >> 4) * 16;

    v8f acc[4][4];
#pragma unroll
    for (int mi = 0; mi < 4; ++mi)
#pragma unroll
        for (int ni = 0; ni < 4; ++ni) { v8f z = {}; acc[mi][ni] = z; }

    // Issue async DMA for one K-slab into buffer `buf`.
    auto stage = [&](int buf, int k0) {
#pragma unroll
        for (int it = 0; it < 2; ++it) {            // A: 128x32 = 512 chunks
            int c = tid + it * 256;
            int r = c >> 2, off = (c & 3) * 8;
            async_ld_b128((unsigned)(size_t)&As[buf][r][off],
                          &A[(size_t)(m0 + r) * K + k0 + off]);
        }
#pragma unroll
        for (int it = 0; it < 4; ++it) {            // B: 256x32 = 1024 chunks
            int c = tid + it * 256;
            int r = c >> 2, off = (c & 3) * 8;
            async_ld_b128((unsigned)(size_t)&Bs[buf][r][off],
                          &Wt[(size_t)(n0 + r) * K + k0 + off]);
        }
    };

    const int nsteps = K >> 5;
    stage(0, 0);                                    // prologue

    for (int s = 0; s < nsteps; ++s) {
        const int buf = s & 1;
        wait_async0();                              // this wave's slab landed
        __syncthreads();                            // every wave's slab landed
        if (s + 1 < nsteps) stage(buf ^ 1, (s + 1) << 5);  // overlap next DMA

        v16h af[4], bf[4];
#pragma unroll
        for (int mi = 0; mi < 4; ++mi)
            af[mi] = *(const v16h*)&As[buf][wm * 64 + mi * 16 + lr16][khalf];
#pragma unroll
        for (int ni = 0; ni < 4; ++ni)
            bf[ni] = *(const v16h*)&Bs[buf][wn * 64 + ni * 16 + lr16][khalf];

#pragma unroll
        for (int mi = 0; mi < 4; ++mi)
#pragma unroll
            for (int ni = 0; ni < 4; ++ni)
                acc[mi][ni] = __builtin_amdgcn_wmma_f32_16x16x32_f16(
                    false, af[mi], false, bf[ni],
                    (short)0, acc[mi][ni], false, false);
        // Buffer `buf` is only re-filled at step s+2's stage(), which all
        // waves reach only after passing step s+1's barrier -> no hazard,
        // no trailing barrier needed here.
    }

    // C/D layout: lanes 0-15 -> M = v, lanes 16-31 -> M = 8+v ; N = lane&15
#pragma unroll
    for (int mi = 0; mi < 4; ++mi)
#pragma unroll
        for (int ni = 0; ni < 4; ++ni) {
            int rb = m0 + wm * 64 + mi * 16 + (lane >> 4) * 8;
            int cc = n0 + wn * 64 + ni * 16 + lr16;
#pragma unroll
            for (int v = 0; v < 8; ++v)
                Cmat[(size_t)(rb + v) * Nn + cc] = acc[mi][ni][v];
        }
}

// ---------------------------------------------------------------------------
// Per-head k/v post-processing: kk-normalize, k_a blend, v_first lerp,
// produce scan vectors a = -kk, b = kk * a_sig.  One 64-thread block per head.
// ---------------------------------------------------------------------------
__global__ __launch_bounds__(64) void post_kv_kernel(
    float* __restrict__ Kbuf, float* __restrict__ Vbuf,
    const float* __restrict__ vfirst, const float* __restrict__ Asig,
    const float* __restrict__ Vgate,
    const float* __restrict__ k_k, const float* __restrict__ k_a,
    float* __restrict__ Ah, float* __restrict__ Bh)
{
    __shared__ float red[64];
    const int row = blockIdx.x / H_;
    const int h   = blockIdx.x % H_;
    const int i   = threadIdx.x;
    const int c   = h * N_ + i;
    const size_t base = (size_t)row * C_ + c;

    float kv  = Kbuf[base];
    float kkc = kv * k_k[c];
    red[i] = kkc * kkc;
    __syncthreads();
    for (int s = 32; s > 0; s >>= 1) { if (i < s) red[i] += red[i + s]; __syncthreads(); }
    float nrm = sqrtf(red[0]);
    float kkn = kkc / fmaxf(nrm, 1e-12f);

    float av = Asig[base];
    float vr = Vbuf[base];
    Kbuf[base] = kv * (1.0f + (av - 1.0f) * k_a[c]);
    Vbuf[base] = vr + (vfirst[base] - vr) * Vgate[base];
    Ah[base]   = -kkn;
    Bh[base]   = kkn * av;
}

// ---------------------------------------------------------------------------
// Sequential RWKV7 scan. One 64-thread block per (b,h); thread i owns
// row i of the 64x64 state in registers; per-step vectors broadcast via LDS.
// ---------------------------------------------------------------------------
__global__ __launch_bounds__(64) void scan_kernel(
    const float* __restrict__ R,  const float* __restrict__ Kh,
    const float* __restrict__ Vh, const float* __restrict__ EW,
    const float* __restrict__ Ah, const float* __restrict__ Bh,
    const float* __restrict__ s0, float* __restrict__ O,
    float* __restrict__ state_out)
{
    __shared__ float Lr[N_], Lk[N_], Lw[N_], La[N_], Lb[N_];
    const int b = blockIdx.x / H_;
    const int h = blockIdx.x % H_;
    const int i = threadIdx.x;

    float S[N_];
    const float* sp = s0 + ((size_t)(b * H_ + h) * N_ + i) * N_;
#pragma unroll
    for (int j = 0; j < N_; ++j) S[j] = sp[j];

    for (int t = 0; t < T_; ++t) {
        const size_t base = (size_t)(b * T_ + t) * C_ + h * N_;
        Lr[i] = R [base + i];
        Lk[i] = Kh[base + i];
        Lw[i] = EW[base + i];
        La[i] = Ah[base + i];
        Lb[i] = Bh[base + i];
        float vi = Vh[base + i];
        __syncthreads();

        float sa = 0.f;
#pragma unroll
        for (int j = 0; j < N_; ++j) sa = fmaf(S[j], La[j], sa);

        float oi = 0.f;
#pragma unroll
        for (int j = 0; j < N_; ++j) {
            float s = fmaf(S[j], Lw[j], fmaf(sa, Lb[j], vi * Lk[j]));
            S[j] = s;
            oi = fmaf(s, Lr[j], oi);
        }
        O[base + i] = oi;
        __syncthreads();
    }

    float* so = state_out + ((size_t)(b * H_ + h) * N_ + i) * N_;
#pragma unroll
    for (int j = 0; j < N_; ++j) so[j] = S[j];
}

// ---------------------------------------------------------------------------
// Per-head groupnorm + bonus + g gating -> f16 operand for final GEMM.
// ---------------------------------------------------------------------------
__global__ __launch_bounds__(64) void post_out_kernel(
    const float* __restrict__ O,  const float* __restrict__ R,
    const float* __restrict__ Kh, const float* __restrict__ Vh,
    const float* __restrict__ r_k, const float* __restrict__ ln_w,
    const float* __restrict__ ln_b, const float* __restrict__ G,
    h16* __restrict__ YG)
{
    __shared__ float red[64];
    const int row = blockIdx.x / H_;
    const int h   = blockIdx.x % H_;
    const int i   = threadIdx.x;
    const int c   = h * N_ + i;
    const size_t base = (size_t)row * C_ + c;

    float oi = O[base];

    red[i] = oi; __syncthreads();
    for (int s = 32; s > 0; s >>= 1) { if (i < s) red[i] += red[i + s]; __syncthreads(); }
    float mean = red[0] * (1.0f / N_);
    __syncthreads();

    red[i] = oi * oi; __syncthreads();
    for (int s = 32; s > 0; s >>= 1) { if (i < s) red[i] += red[i + s]; __syncthreads(); }
    float var = red[0] * (1.0f / N_) - mean * mean;
    __syncthreads();

    red[i] = R[base] * Kh[base] * r_k[c]; __syncthreads();
    for (int s = 32; s > 0; s >>= 1) { if (i < s) red[i] += red[i + s]; __syncthreads(); }
    float dot = red[0];

    float on = (oi - mean) * rsqrtf(var + EPSV) * ln_w[c] + ln_b[c];
    float y  = on + dot * Vh[base];
    YG[base] = (h16)(y * G[base]);
}

// ---------------------------------------------------------------------------
// Launch
// ---------------------------------------------------------------------------
extern "C" void kernel_launch(void* const* d_in, const int* in_sizes, int n_in,
                              void* d_out, int out_size, void* d_ws, size_t ws_size,
                              hipStream_t stream)
{
    const float* x      = (const float*)d_in[0];
    const float* vfirst = (const float*)d_in[1];
    const float* shift  = (const float*)d_in[2];
    const float* wkv0   = (const float*)d_in[3];
    const float* xr_m = (const float*)d_in[4];
    const float* xw_m = (const float*)d_in[5];
    const float* xk_m = (const float*)d_in[6];
    const float* xv_m = (const float*)d_in[7];
    const float* xa_m = (const float*)d_in[8];
    const float* xg_m = (const float*)d_in[9];
    const float* w0 = (const float*)d_in[10];
    const float* w1 = (const float*)d_in[11];
    const float* w2 = (const float*)d_in[12];
    const float* a0 = (const float*)d_in[13];
    const float* a1 = (const float*)d_in[14];
    const float* a2 = (const float*)d_in[15];
    const float* v0 = (const float*)d_in[16];
    const float* v1 = (const float*)d_in[17];
    const float* v2 = (const float*)d_in[18];
    const float* g1 = (const float*)d_in[19];
    const float* g2 = (const float*)d_in[20];
    const float* k_k = (const float*)d_in[21];
    const float* k_a = (const float*)d_in[22];
    const float* r_k = (const float*)d_in[23];
    const float* W_r = (const float*)d_in[24];
    const float* W_k = (const float*)d_in[25];
    const float* W_v = (const float*)d_in[26];
    const float* W_o = (const float*)d_in[27];
    const float* ln_w = (const float*)d_in[28];
    const float* ln_b = (const float*)d_in[29];

    const size_t BTC = (size_t)BT_ * C_;
    const size_t CC  = (size_t)C_ * C_;

    float* out        = (float*)d_out;              // (B,T,C)
    float* vfirst_out = out + BTC;                  // (B,T,C)
    float* lx_out     = out + 2 * BTC;              // (B,C)
    float* state_out  = lx_out + (size_t)B_ * C_;   // (B,H,N,N)

    // Carve workspace
    char* wsp = (char*)d_ws;
    auto carve = [&](size_t bytes) -> void* {
        void* p = (void*)wsp;
        wsp += (bytes + 255) & ~(size_t)255;
        return p;
    };
    h16* Xr = (h16*)carve(BTC * 2);
    h16* Xk = (h16*)carve(BTC * 2);
    h16* Xv = (h16*)carve(BTC * 2);
    h16* Xg = (h16*)carve(BTC * 2);
    h16* Wr_h = (h16*)carve(CC * 2);
    h16* Wk_h = (h16*)carve(CC * 2);
    h16* Wv_h = (h16*)carve(CC * 2);
    h16* Wo_h = (h16*)carve(CC * 2);
    float* Rb   = (float*)carve(BTC * 4);
    float* Kb   = (float*)carve(BTC * 4);
    float* Vb   = (float*)carve(BTC * 4);
    float* EW   = (float*)carve(BTC * 4);
    float* Asig = (float*)carve(BTC * 4);
    float* Vg   = (float*)carve(BTC * 4);
    float* Gb   = (float*)carve(BTC * 4);
    float* Ahb  = (float*)carve(BTC * 4);
    float* Bhb  = (float*)carve(BTC * 4);
    float* Ob   = (float*)carve(BTC * 4);
    h16* YG = Xr;   // Xr is dead after the R GEMM -> reuse for gated output

    // 1) weights -> f16
    int cvblk = (int)((CC + 255) / 256);
    f32_to_f16_kernel<<<cvblk, 256, 0, stream>>>(W_r, Wr_h, CC);
    f32_to_f16_kernel<<<cvblk, 256, 0, stream>>>(W_k, Wk_h, CC);
    f32_to_f16_kernel<<<cvblk, 256, 0, stream>>>(W_v, Wv_h, CC);
    f32_to_f16_kernel<<<cvblk, 256, 0, stream>>>(W_o, Wo_h, CC);

    // 2) token shift + mixes + LoRA MLPs (+ v_first copy, lx)
    mix_mlp_kernel<<<BT_, 256, 0, stream>>>(
        x, vfirst, shift, xr_m, xw_m, xk_m, xv_m, xa_m, xg_m,
        w0, w1, w2, a0, a1, a2, v0, v1, v2, g1, g2,
        Xr, Xk, Xv, Xg, EW, Asig, Vg, Gb, vfirst_out, lx_out);

    // 3) big WMMA GEMMs: r, k, v   (block tile 128 x 256)
    dim3 gg(C_ / 256, BT_ / 128);
    gemm_f16_wmma<<<gg, 256, 0, stream>>>(Xr, Wr_h, Rb, BT_, C_, C_);
    gemm_f16_wmma<<<gg, 256, 0, stream>>>(Xk, Wk_h, Kb, BT_, C_, C_);
    gemm_f16_wmma<<<gg, 256, 0, stream>>>(Xv, Wv_h, Vb, BT_, C_, C_);

    // 4) per-head post-processing of k, v -> scan operands
    post_kv_kernel<<<BT_ * H_, 64, 0, stream>>>(Kb, Vb, vfirst, Asig, Vg,
                                                k_k, k_a, Ahb, Bhb);

    // 5) sequential scan (+ final state output)
    scan_kernel<<<B_ * H_, 64, 0, stream>>>(Rb, Kb, Vb, EW, Ahb, Bhb,
                                            wkv0, Ob, state_out);

    // 6) groupnorm + bonus + g gating -> f16
    post_out_kernel<<<BT_ * H_, 64, 0, stream>>>(Ob, Rb, Kb, Vb, r_k,
                                                 ln_w, ln_b, Gb, YG);

    // 7) final WMMA GEMM straight into d_out
    gemm_f16_wmma<<<gg, 256, 0, stream>>>(YG, Wo_h, out, BT_, C_, C_);
}
